// Attention_74852690035272
// MI455X (gfx1250) — compile-verified, matrix operand order checked by reading, not run on hardware
//
#include <hip/hip_runtime.h>

typedef unsigned short u16;
typedef unsigned int u32;
typedef unsigned long long u64;

typedef __attribute__((ext_vector_type(16))) __bf16 v16bf;
typedef __attribute__((ext_vector_type(8)))  float  v8f;

#define BATCH 2
#define SEQ   2048
#define DIM   1024
#define HEADS 16
#define DH    64
#define INNER 1024
#define ATTN_SCALE 0.125f   /* 64^-0.5 */

union Frag { v16bf v; u32 u[8]; };

__device__ __forceinline__ u16 f2bf(float f) {
    union { float f; u32 u; } x; x.f = f;
    u32 r = x.u + 0x7fffu + ((x.u >> 16) & 1u);
    return (u16)(r >> 16);
}
__device__ __forceinline__ u32 pack2(u16 lo, u16 hi) {
    return (u32)lo | ((u32)hi << 16);
}

// A/B fragment K-pair pattern for 16-bit 16x16x32 WMMA (wave32):
// lanes 0-15: K = {0..7, 16..23}; lanes 16-31: K = {8..15, 24..31}, packed 2/VGPR.
__device__ __forceinline__ int kpat(int j, int hi) {
    return ((j < 4) ? (hi ? 8 : 0) : (hi ? 24 : 16)) + 2 * (j & 3);
}

__device__ __forceinline__ void ld_frag(Frag& f, const u16* p, int stride, int row, int hi) {
#pragma unroll
    for (int j = 0; j < 8; ++j)
        f.u[j] = *(const u32*)(p + row * stride + kpat(j, hi));
}

__device__ __forceinline__ v8f wmma_bf16(const Frag& a, const Frag& b, v8f c) {
    return __builtin_amdgcn_wmma_f32_16x16x32_bf16(false, a.v, false, b.v,
                                                   (short)0, c, false, false);
}

__device__ __forceinline__ float rmax16(float v) {
#pragma unroll
    for (int m = 8; m >= 1; m >>= 1) v = fmaxf(v, __shfl_xor(v, m, 32));
    return v;
}
__device__ __forceinline__ float rsum16(float v) {
#pragma unroll
    for (int m = 8; m >= 1; m >>= 1) v += __shfl_xor(v, m, 32);
    return v;
}

// ---------------------------------------------------------------------------
// Tensor Data Mover: 2D bf16 tile copy global -> LDS with hardware row
// padding (1 DWORD pad every 2^(pad_code+1) DWORDs -> matches the +2 u16 LDS
// row padding used below).  D# layout per CDNA5 ISA.
// 6-arg builtin form (clang-23 / therock-10.0); manual-copy fallback.
// ---------------------------------------------------------------------------
#if defined(__gfx1250__) && __has_builtin(__builtin_amdgcn_tensor_load_to_lds)
#define USE_TDM 1
typedef __attribute__((ext_vector_type(4))) unsigned int v4u_;
typedef __attribute__((ext_vector_type(8))) int v8i_;
typedef __attribute__((ext_vector_type(4))) int v4i_;

__device__ __forceinline__ void tdm_load_2d(u32 lds_byte_addr, const void* gaddr,
                                            u32 tile_d0, u32 tile_d1,
                                            u32 stride0_elems, u32 pad_code)
{
    const u64 ga = (u64)gaddr;
    v4u_ g0;
    g0.x = 1u;                                            // count=1, user D#
    g0.y = lds_byte_addr;                                 // lds_addr
    g0.z = (u32)ga;                                       // global_addr[31:0]
    g0.w = (u32)((ga >> 32) & 0x01FFFFFFu) | (2u << 30);  // addr[56:32], type=2
    v8i_ g1;
    g1[0] = (int)((1u << 16) | (1u << 20) | (pad_code << 22)); // 2B elems, pad_en, interval; pad=1 DW
    g1[1] = (int)((tile_d0 & 0xFFFFu) << 16);             // tensor_dim0[15:0]
    g1[2] = (int)((tile_d1 & 0xFFFFu) << 16);             // tensor_dim1[15:0] (dim0 hi = 0)
    g1[3] = (int)((tile_d0 & 0xFFFFu) << 16);             // tile_dim0 (dim1 hi = 0)
    g1[4] = (int)(tile_d1 & 0xFFFFu);                     // tile_dim1 (tile_dim2 = 0)
    g1[5] = (int)stride0_elems;                           // tensor_dim0_stride[31:0]
    g1[6] = 0;
    g1[7] = 0;
    const v4i_ z4 = {0, 0, 0, 0};
    const v8i_ z8 = {0, 0, 0, 0, 0, 0, 0, 0};
    __builtin_amdgcn_tensor_load_to_lds(g0, g1, z4, z4, z8, 0);
}
__device__ __forceinline__ void tdm_wait() {
    __builtin_amdgcn_s_wait_tensorcnt(0);
}
#else
#define USE_TDM 0
#endif

// ---------------------------------------------------------------------------
// Kernel 1: fused QKV projection (f32 -> bf16 WMMA).  Q,K written [B][H][S][DH];
// V written TRANSPOSED [B][H][DH][S] so attention V-chunks are plain 2D tiles.
// ---------------------------------------------------------------------------
__global__ __launch_bounds__(256) void qkv_proj_kernel(
    const float* __restrict__ x,
    const float* __restrict__ Wq, const float* __restrict__ Wk,
    const float* __restrict__ Wv,
    u16* __restrict__ q, u16* __restrict__ k, u16* __restrict__ v)
{
    const float* W;
    u16* o;
    if (blockIdx.z == 0)      { W = Wq; o = q; }
    else if (blockIdx.z == 1) { W = Wk; o = k; }
    else                      { W = Wv; o = v; }
    const bool vtrans = (blockIdx.z == 2);

    const int m0 = blockIdx.x * 128;
    const int n0 = blockIdx.y * 64;

    __shared__ u16 As[128][34];
    __shared__ u16 Bs[64][34];

    const int tid = threadIdx.x, lane = tid & 31, wave = tid >> 5;
    const int hi = lane >> 4, ln = lane & 15;
    const int wm = (wave & 3) * 32, wn = (wave >> 2) * 32;

    v8f acc[2][2];
#pragma unroll
    for (int i = 0; i < 2; ++i)
#pragma unroll
        for (int j = 0; j < 2; ++j)
            acc[i][j] = (v8f){0.f,0.f,0.f,0.f,0.f,0.f,0.f,0.f};

    for (int kk = 0; kk < DIM; kk += 32) {
        {   // stage A: 128x32, convert f32 -> bf16
            const int row = tid >> 1, c0 = (tid & 1) * 16;
            const float* src = x + (size_t)(m0 + row) * DIM + kk + c0;
            __builtin_prefetch(src + 32, 0, 1);
#pragma unroll
            for (int i = 0; i < 16; i += 2)
                *(u32*)&As[row][c0 + i] = pack2(f2bf(src[i]), f2bf(src[i + 1]));
        }
        {   // stage B transposed: W[kk+kr][n0+nb..] -> Bs[n][k]
            const int kr = tid >> 3, nb = (tid & 7) * 8;
            const float* src = W + (size_t)(kk + kr) * DIM + n0 + nb;
#pragma unroll
            for (int i = 0; i < 8; ++i)
                Bs[nb + i][kr] = f2bf(src[i]);
        }
        __syncthreads();

        Frag a[2], b[2];
#pragma unroll
        for (int mi = 0; mi < 2; ++mi) ld_frag(a[mi], &As[0][0], 34, wm + mi * 16 + ln, hi);
#pragma unroll
        for (int ni = 0; ni < 2; ++ni) ld_frag(b[ni], &Bs[0][0], 34, wn + ni * 16 + ln, hi);
#pragma unroll
        for (int mi = 0; mi < 2; ++mi)
#pragma unroll
            for (int ni = 0; ni < 2; ++ni)
                acc[mi][ni] = wmma_bf16(a[mi], b[ni], acc[mi][ni]);
        __syncthreads();
    }

#pragma unroll
    for (int mi = 0; mi < 2; ++mi)
#pragma unroll
        for (int ni = 0; ni < 2; ++ni)
#pragma unroll
            for (int r = 0; r < 8; ++r) {
                const int row = m0 + wm + mi * 16 + r + hi * 8;
                const int col = n0 + wn + ni * 16 + ln;
                const int bb = row >> 11, ss = row & (SEQ - 1);
                const int hh = col >> 6,  dd = col & (DH - 1);
                const size_t base = ((size_t)bb * HEADS + hh);
                const size_t idx = vtrans ? (base * DH + dd) * SEQ + ss
                                          : (base * SEQ + ss) * DH + dd;
                o[idx] = f2bf(acc[mi][ni][r]);
            }
}

// ---------------------------------------------------------------------------
// Kernel 2: flash attention.  One block = 64 query rows of one (b,h); 4 waves.
// K/V chunks stream through double-buffered LDS via the Tensor Data Mover
// (wave 0 issues, s_wait_tensorcnt, one barrier per chunk).  Masked logits
// contribute exactly 0 to the exp-sum; final divide by max(l, 1e-9).
// ---------------------------------------------------------------------------
__global__ __launch_bounds__(128) void attn_kernel(
    const u16* __restrict__ q, const u16* __restrict__ k,
    const u16* __restrict__ v, const unsigned char* __restrict__ mask,
    u16* __restrict__ ao)
{
    const int b = blockIdx.z, h = blockIdx.y, q0 = blockIdx.x * 64;
    const size_t bh = ((size_t)b * HEADS + h);

    __shared__ u16 Ks[2][64][66];    // [buf][j][d]  (sim B-frag: K-dim = d)
    __shared__ u16 Vs[2][64][66];    // [buf][d][j]  (PV  B-frag: K-dim = j)
    __shared__ u16 Ps[4][16][66];    // per-wave P staging [i][j]

    const int tid = threadIdx.x, lane = tid & 31, wave = tid >> 5;
    const int hi = lane >> 4, ln = lane & 15;

    const u16* kbase  = k + bh * SEQ * DH;   // [S][DH]
    const u16* vtbase = v + bh * DH * SEQ;   // [DH][S]  (transposed V)

#if USE_TDM
    if (wave == 0) {   // pre-issue chunk 0 (overlaps with Q fragment loads)
        tdm_load_2d((u32)(size_t)&Ks[0][0][0], kbase,  64, 64, DH,  4);
        tdm_load_2d((u32)(size_t)&Vs[0][0][0], vtbase, 64, 64, SEQ, 4);
    }
#endif

    // Q fragments (held in registers for the whole row-block)
    Frag qa[2];
    {
        const int qrow = q0 + wave * 16 + ln;
        const u16* qp = q + (bh * SEQ + qrow) * DH;
#pragma unroll
        for (int f = 0; f < 2; ++f) ld_frag(qa[f], qp + f * 32, 0, 0, hi);
    }

    v8f O[4];
#pragma unroll
    for (int dt = 0; dt < 4; ++dt) O[dt] = (v8f){0.f,0.f,0.f,0.f,0.f,0.f,0.f,0.f};
    float mr[8], lr[8];
#pragma unroll
    for (int r = 0; r < 8; ++r) { mr[r] = -3.0e38f; lr[r] = 0.f; }

    for (int c = 0; c < SEQ / 64; ++c) {
        const int cur = c & 1;
        const int j0 = c * 64;
#if USE_TDM
        if (wave == 0) tdm_wait();          // current buffers resident
        __syncthreads();                    // release consumers; prev reads done
        if (wave == 0 && c + 1 < SEQ / 64) {
            const int jn = j0 + 64;         // prefetch next chunk into other buf
            tdm_load_2d((u32)(size_t)&Ks[cur ^ 1][0][0], kbase + (size_t)jn * DH,
                        64, 64, DH, 4);
            tdm_load_2d((u32)(size_t)&Vs[cur ^ 1][0][0], vtbase + jn,
                        64, 64, SEQ, 4);
        }
#else
        {   // manual cooperative tile copy (both sources row-contiguous)
            const int rr = tid >> 1, c0 = (tid & 1) * 32;
            const u16* ksrc = kbase  + (size_t)(j0 + rr) * DH + c0;
            const u16* vsrc = vtbase + (size_t)rr * SEQ + j0 + c0;
#pragma unroll
            for (int i = 0; i < 32; i += 2) {
                *(u32*)&Ks[cur][rr][c0 + i] = *(const u32*)(ksrc + i);
                *(u32*)&Vs[cur][rr][c0 + i] = *(const u32*)(vsrc + i);
            }
        }
        __syncthreads();
#endif

        // ---- sim = Q . K^T over this chunk (4 tiles of 16x16), mask+scale ----
        float sv[4][8];
#pragma unroll
        for (int jt = 0; jt < 4; ++jt) {
            Frag kb0, kb1;
            ld_frag(kb0, &Ks[cur][0][0],      66, jt * 16 + ln, hi);  // d 0..31
            ld_frag(kb1, &Ks[cur][0][0] + 32, 66, jt * 16 + ln, hi);  // d 32..63
            v8f s = (v8f){0.f,0.f,0.f,0.f,0.f,0.f,0.f,0.f};
            s = wmma_bf16(qa[0], kb0, s);
            s = wmma_bf16(qa[1], kb1, s);
            const int jcol = j0 + jt * 16 + ln;
#pragma unroll
            for (int r = 0; r < 8; ++r) {
                const int irow = q0 + wave * 16 + r + hi * 8;
                const unsigned char mk = mask[((size_t)b * SEQ + irow) * SEQ + jcol];
                sv[jt][r] = mk ? s[r] * ATTN_SCALE : -1.0e30f;
            }
        }

        // ---- online softmax update ----
        float mnew[8];
#pragma unroll
        for (int r = 0; r < 8; ++r) {
            float x = fmaxf(fmaxf(sv[0][r], sv[1][r]), fmaxf(sv[2][r], sv[3][r]));
            mnew[r] = fmaxf(mr[r], rmax16(x));
        }
#pragma unroll
        for (int r = 0; r < 8; ++r) {
            const float sc = __expf(mr[r] - mnew[r]);
            lr[r] *= sc;
#pragma unroll
            for (int dt = 0; dt < 4; ++dt) O[dt][r] *= sc;
            mr[r] = mnew[r];
        }
        float psum[8];
#pragma unroll
        for (int r = 0; r < 8; ++r) psum[r] = 0.f;
#pragma unroll
        for (int jt = 0; jt < 4; ++jt)
#pragma unroll
            for (int r = 0; r < 8; ++r) {
                const float p = (sv[jt][r] > -1.0e29f) ? __expf(sv[jt][r] - mr[r]) : 0.f;
                psum[r] += p;
                Ps[wave][r + hi * 8][jt * 16 + ln] = f2bf(p);   // C-layout -> A-layout staging
            }
#pragma unroll
        for (int r = 0; r < 8; ++r) lr[r] += rsum16(psum[r]);

        // ---- O += P @ V  (per-wave LDS round-trip; DS ops in-order per wave) ----
        Frag pa[2];
#pragma unroll
        for (int f = 0; f < 2; ++f) ld_frag(pa[f], &Ps[wave][0][0] + f * 32, 66, ln, hi);
#pragma unroll
        for (int dt = 0; dt < 4; ++dt) {
            Frag vb0, vb1;
            ld_frag(vb0, &Vs[cur][0][0],      66, dt * 16 + ln, hi);  // j 0..31
            ld_frag(vb1, &Vs[cur][0][0] + 32, 66, dt * 16 + ln, hi);  // j 32..63
            O[dt] = wmma_bf16(pa[0], vb0, O[dt]);
            O[dt] = wmma_bf16(pa[1], vb1, O[dt]);
        }
#if !USE_TDM
        __syncthreads();
#endif
    }

    // ---- finalize and write attn_out [B*S][INNER] bf16 ----
#pragma unroll
    for (int r = 0; r < 8; ++r) {
        const float inv = 1.f / fmaxf(lr[r], 1e-9f);
        const int irow = q0 + wave * 16 + r + hi * 8;
#pragma unroll
        for (int dt = 0; dt < 4; ++dt) {
            const int col = dt * 16 + ln;
            ao[((size_t)b * SEQ + irow) * INNER + h * DH + col] =
                f2bf(O[dt][r] * inv);
        }
    }
}

// ---------------------------------------------------------------------------
// Kernel 3: output projection.  out = attn_out(bf16) @ Wo + bo  (f32 out).
// A-tile (raw bf16 copy) staged via TDM; B-tile converted f32->bf16 by VALU.
// ---------------------------------------------------------------------------
__global__ __launch_bounds__(256) void out_proj_kernel(
    const u16* __restrict__ ao, const float* __restrict__ Wo,
    const float* __restrict__ bo, float* __restrict__ out)
{
    const int m0 = blockIdx.x * 128;
    const int n0 = blockIdx.y * 64;

    __shared__ u16 As[128][34];
    __shared__ u16 Bs[64][34];

    const int tid = threadIdx.x, lane = tid & 31, wave = tid >> 5;
    const int hi = lane >> 4, ln = lane & 15;
    const int wm = (wave & 3) * 32, wn = (wave >> 2) * 32;

    v8f acc[2][2];
#pragma unroll
    for (int i = 0; i < 2; ++i)
#pragma unroll
        for (int j = 0; j < 2; ++j)
            acc[i][j] = (v8f){0.f,0.f,0.f,0.f,0.f,0.f,0.f,0.f};

    for (int kk = 0; kk < INNER; kk += 32) {
#if USE_TDM
        if (wave == 0)   // 128x32 bf16 tile, row stride 1024, pad to 34-u16 rows
            tdm_load_2d((u32)(size_t)&As[0][0], ao + (size_t)m0 * INNER + kk,
                        32, 128, INNER, 3);
#else
        {
            const int row = tid >> 1, c0 = (tid & 1) * 16;
            const u16* src = ao + (size_t)(m0 + row) * INNER + kk + c0;
#pragma unroll
            for (int i = 0; i < 16; i += 2)
                *(u32*)&As[row][c0 + i] = *(const u32*)(src + i);
        }
#endif
        {   // stage B transposed, convert f32 -> bf16 (overlaps TDM)
            const int kr = tid >> 3, nb = (tid & 7) * 8;
            const float* src = Wo + (size_t)(kk + kr) * DIM + n0 + nb;
#pragma unroll
            for (int i = 0; i < 8; ++i)
                Bs[nb + i][kr] = f2bf(src[i]);
        }
#if USE_TDM
        if (wave == 0) tdm_wait();
#endif
        __syncthreads();

        Frag a[2], b[2];
#pragma unroll
        for (int mi = 0; mi < 2; ++mi) ld_frag(a[mi], &As[0][0], 34, wm + mi * 16 + ln, hi);
#pragma unroll
        for (int ni = 0; ni < 2; ++ni) ld_frag(b[ni], &Bs[0][0], 34, wn + ni * 16 + ln, hi);
#pragma unroll
        for (int mi = 0; mi < 2; ++mi)
#pragma unroll
            for (int ni = 0; ni < 2; ++ni)
                acc[mi][ni] = wmma_bf16(a[mi], b[ni], acc[mi][ni]);
        __syncthreads();
    }

#pragma unroll
    for (int mi = 0; mi < 2; ++mi)
#pragma unroll
        for (int ni = 0; ni < 2; ++ni)
#pragma unroll
            for (int r = 0; r < 8; ++r) {
                const int row = m0 + wm + mi * 16 + r + hi * 8;
                const int col = n0 + wn + ni * 16 + ln;
                out[(size_t)row * DIM + col] = acc[mi][ni][r] + bo[col];
            }
}

// ---------------------------------------------------------------------------
extern "C" void kernel_launch(void* const* d_in, const int* in_sizes, int n_in,
                              void* d_out, int out_size, void* d_ws, size_t ws_size,
                              hipStream_t stream) {
    const float* x  = (const float*)d_in[0];
    const unsigned char* mask = (const unsigned char*)d_in[1];
    const float* Wq = (const float*)d_in[2];
    const float* Wk = (const float*)d_in[3];
    const float* Wv = (const float*)d_in[4];
    const float* Wo = (const float*)d_in[5];
    const float* bo = (const float*)d_in[6];

    const size_t QKV = (size_t)BATCH * SEQ * INNER;   // 4,194,304 elements
    u16* q  = (u16*)d_ws;
    u16* k  = q + QKV;
    u16* v  = k + QKV;     // stored [B][H][DH][S]
    u16* ao = v + QKV;     // total 32 MB of ws

    qkv_proj_kernel<<<dim3(BATCH * SEQ / 128, INNER / 64, 3), 256, 0, stream>>>(
        x, Wq, Wk, Wv, q, k, v);
    attn_kernel<<<dim3(SEQ / 64, HEADS, BATCH), 128, 0, stream>>>(
        q, k, v, mask, ao);
    out_proj_kernel<<<dim3(BATCH * SEQ / 128, DIM / 64), 256, 0, stream>>>(
        ao, Wo, bo, (float*)d_out);
}
